// DMGHAN_25993142075977
// MI455X (gfx1250) — matrix operand order repaired and checked by reading, not compiled, not run on hardware
//
#include <hip/hip_runtime.h>
#include <hip/hip_bf16.h>
#include <cstddef>

#define BB 8
#define LL 2048
#define EMBED 1024
#define DMODEL 256
#define DINNER 512
#define DSTATE 16
#define DTRANK 16
#define MTOT (BB * LL)   // 16384

typedef __attribute__((ext_vector_type(16))) __bf16 bf16x16;
typedef __attribute__((ext_vector_type(8)))  __bf16 bf16x8;
typedef __attribute__((ext_vector_type(4)))  __bf16 bf16x4;
typedef __attribute__((ext_vector_type(8)))  float  f32x8;
typedef __attribute__((ext_vector_type(4)))  float  f32x4;

static __device__ __forceinline__ float siluf(float x) { return x / (1.0f + __expf(-x)); }
static __device__ __forceinline__ float softplusf(float x) { return (x > 20.0f) ? x : log1pf(__expf(x)); }

// ---------------------------------------------------------------------------
// WMMA GEMM: C[M,N] = act( A[M,K](lda) * W[N,K]^T + bias )
// Block tile 128x64xK32, 256 threads (8 waves). Wave w owns rows [w*16,w*16+16)
// and all four 16-col subtiles -> 4 v_wmma per K-step, A-fragment reused 4x.
// RAGGED=false: M%128==0, N%64==0, K%32==0 guaranteed -> unguarded loads.
// HAS_BIAS / ACT compile-time to keep the epilogue branch-free.
// ---------------------------------------------------------------------------
template<bool RAGGED, bool HAS_BIAS, int ACT>
__global__ __launch_bounds__(256) void gemm_wmma_bf16(
    const float* __restrict__ A, int lda,
    const float* __restrict__ W,
    const float* __restrict__ bias,
    float* __restrict__ C, int ldc,
    int M, int N, int K)
{
    __shared__ alignas(64) __bf16 As[128][32];
    __shared__ alignas(64) __bf16 Bs[64][32];

    const int bm = blockIdx.x * 128;
    const int bn = blockIdx.y * 64;
    const int tid = threadIdx.x;
    const int lane = tid & 31;
    const int w = tid >> 5;

    const int sm  = w * 16;          // wave's M sub-offset in tile
    const int l15 = lane & 15;
    const int kh  = lane >> 4;

    f32x8 acc[4] = {{}, {}, {}, {}};

    const int kt = (K + 31) / 32;
    for (int t = 0; t < kt; ++t) {
        const int k0 = t * 32;
        // ---- stage A tile: 128x32 fp32 -> bf16 (1024 float4, 4 per thread)
        #pragma unroll
        for (int i = 0; i < 4; ++i) {
            int idx = tid + i * 256;
            int r   = idx >> 3;
            int c4  = (idx & 7) * 4;
            int gm = bm + r, gk = k0 + c4;
            f32x4 v;
            if (!RAGGED) {
                v = *(const f32x4*)(A + (size_t)gm * lda + gk);
            } else {
                v = (f32x4){0.f, 0.f, 0.f, 0.f};
                if (gk + 4 <= K) {
                    v = *(const f32x4*)(A + (size_t)gm * lda + gk);
                } else {
                    for (int j = 0; j < 4; ++j) if (gk + j < K) v[j] = A[(size_t)gm * lda + gk + j];
                }
            }
            *(bf16x4*)&As[r][c4] = __builtin_convertvector(v, bf16x4);
        }
        // ---- stage W tile: 64x32, N-major (512 float4, 2 per thread)
        #pragma unroll
        for (int i = 0; i < 2; ++i) {
            int idx = tid + i * 256;
            int r   = idx >> 3;
            int c4  = (idx & 7) * 4;
            int gn = bn + r, gk = k0 + c4;
            f32x4 v;
            if (!RAGGED) {
                v = *(const f32x4*)(W + (size_t)gn * K + gk);
            } else {
                v = (f32x4){0.f, 0.f, 0.f, 0.f};
                if (gn < N) {
                    if (gk + 4 <= K) {
                        v = *(const f32x4*)(W + (size_t)gn * K + gk);
                    } else {
                        for (int j = 0; j < 4; ++j) if (gk + j < K) v[j] = W[(size_t)gn * K + gk + j];
                    }
                }
            }
            *(bf16x4*)&Bs[r][c4] = __builtin_convertvector(v, bf16x4);
        }
        __syncthreads();

        // A fragment: lane holds M=l15; elems 0..7 = K[kh*8..], 8..15 = K[16+kh*8..]
        bf16x8 alo = *(const bf16x8*)&As[sm + l15][kh * 8];
        bf16x8 ahi = *(const bf16x8*)&As[sm + l15][16 + kh * 8];
        bf16x16 af = __builtin_shufflevector(alo, ahi, 0,1,2,3,4,5,6,7,8,9,10,11,12,13,14,15);

        #pragma unroll
        for (int s = 0; s < 4; ++s) {
            // B fragment: lane holds N=l15; elems 0..7 = K[kh*8..], 8..15 = K[16+kh*8..]
            bf16x8 blo = *(const bf16x8*)&Bs[s * 16 + l15][kh * 8];
            bf16x8 bhi = *(const bf16x8*)&Bs[s * 16 + l15][16 + kh * 8];
            bf16x16 bfr = __builtin_shufflevector(blo, bhi, 0,1,2,3,4,5,6,7,8,9,10,11,12,13,14,15);
            acc[s] = __builtin_amdgcn_wmma_f32_16x16x32_bf16(false, af, false, bfr, (short)0, acc[s], false, false);
        }
        __syncthreads();
    }

    // C/D layout: lane N = l15 (lanes 16-31 -> M += 8); VGPR r -> M = r
    const int mbase = bm + sm + kh * 8;
    #pragma unroll
    for (int s = 0; s < 4; ++s) {
        int gn = bn + s * 16 + l15;
        if (RAGGED && gn >= N) continue;
        float bv = HAS_BIAS ? bias[gn] : 0.f;
        #pragma unroll
        for (int r = 0; r < 8; ++r) {
            float v = acc[s][r] + bv;
            if (ACT == 1) v = softplusf(v);
            C[(size_t)(mbase + r) * ldc + gn] = v;
        }
    }
}

// ---------------------------------------------------------------------------
// Causal depthwise conv (taps=4) + bias + silu.  u lives in uz[:, :512].
// ---------------------------------------------------------------------------
__global__ void conv_silu_kernel(const float* __restrict__ uz,
                                 const float* __restrict__ conv_w,
                                 const float* __restrict__ conv_b,
                                 float* __restrict__ ubuf)
{
    size_t i = (size_t)blockIdx.x * blockDim.x + threadIdx.x;
    if (i >= (size_t)MTOT * DINNER) return;
    int d = (int)(i % DINNER);
    size_t bl = i / DINNER;
    int l = (int)(bl % LL);
    int b = (int)(bl / LL);
    const float* up = uz + ((size_t)b * LL) * (2 * DINNER) + d;
    float acc = conv_b[d];
    #pragma unroll
    for (int j = 0; j < 4; ++j) {
        int ll = l - 3 + j;
        if (ll >= 0) acc += up[(size_t)ll * (2 * DINNER)] * conv_w[d * 4 + j];
    }
    ubuf[i] = siluf(acc);
}

// ---------------------------------------------------------------------------
// Selective scan. Thread per (b, d, n): 65536 threads, state in a register.
// B/C chunks staged global->LDS with CDNA5 async-LDS loads (ASYNCcnt path):
// 512 threads x 1 global_load_async_to_lds_b128 covers 64 rows x {B,C}.
// y reduced over n (16 lanes) via shfl_xor; silu(z) gating fused; in-place out.
// ---------------------------------------------------------------------------
#define SCAN_CH 64
__global__ __launch_bounds__(512) void scan_kernel(
    const float* __restrict__ dtbuf, const float* __restrict__ ubuf,
    const float* __restrict__ xdbc,  const float* __restrict__ uzbuf,
    const float* __restrict__ A_log, const float* __restrict__ Dp,
    float* __restrict__ yg)
{
    __shared__ alignas(16) float Bs[SCAN_CH][DSTATE];
    __shared__ alignas(16) float Cs[SCAN_CH][DSTATE];
    const int b  = blockIdx.x >> 4;          // 8 batches
    const int dg = blockIdx.x & 15;          // 16 d-groups of 32
    const int n  = threadIdx.x & 15;
    const int d  = dg * 32 + (threadIdx.x >> 4);

    const float Aa  = -__expf(A_log[d * DSTATE + n]);
    const float Dpd = Dp[d];
    float h = 0.f;

    // per-thread async-copy slot (fixed across chunks)
    const int cidx  = threadIdx.x;           // 0..511
    const int which = cidx >> 8;             // 0 = B, 1 = C
    const int crow  = (cidx >> 2) & 63;
    const int cq    = cidx & 3;
    const unsigned ldsaddr = (unsigned)(unsigned long long)(const void*)
        (which ? &Cs[crow][cq * 4] : &Bs[crow][cq * 4]);

    for (int l0 = 0; l0 < LL; l0 += SCAN_CH) {
        {
            const float* gp = xdbc + ((size_t)b * LL + l0 + crow) * 48
                              + DTRANK + which * DSTATE + cq * 4;
            unsigned long long ga = (unsigned long long)gp;
            asm volatile("global_load_async_to_lds_b128 %0, %1, off"
                         :: "v"(ldsaddr), "v"(ga) : "memory");
        }
        asm volatile("s_wait_asynccnt 0x0" ::: "memory");
        __syncthreads();
        for (int li = 0; li < SCAN_CH; ++li) {
            size_t m = (size_t)b * LL + l0 + li;
            float dt = dtbuf[m * DINNER + d];
            float uu = ubuf [m * DINNER + d];
            h = __expf(dt * Aa) * h + (dt * uu) * Bs[li][n];
            float y = h * Cs[li][n];
            y += __shfl_xor(y, 8, 16);
            y += __shfl_xor(y, 4, 16);
            y += __shfl_xor(y, 2, 16);
            y += __shfl_xor(y, 1, 16);
            if (n == 0) {
                float z = uzbuf[m * (2 * DINNER) + DINNER + d];
                yg[m * DINNER + d] = (y + uu * Dpd) * siluf(z);
            }
        }
        __syncthreads();
    }
}

// ---------------------------------------------------------------------------
// Mean pool over L.
// ---------------------------------------------------------------------------
__global__ void pool_kernel(const float* __restrict__ seq, float* __restrict__ pooled)
{
    int b = blockIdx.x, j = threadIdx.x;
    float s = 0.f;
    for (int l = 0; l < LL; ++l) s += seq[((size_t)b * LL + l) * DMODEL + j];
    pooled[b * DMODEL + j] = s * (1.0f / LL);
}

// ---------------------------------------------------------------------------
// Small dense: C[8,N] = A[8,K] * W[N,K]^T + bias
// ---------------------------------------------------------------------------
__global__ void small_linear_kernel(const float* __restrict__ A,
                                    const float* __restrict__ W,
                                    const float* __restrict__ bias,
                                    float* __restrict__ C, int K, int N)
{
    int idx = blockIdx.x * blockDim.x + threadIdx.x;
    if (idx >= BB * N) return;
    int b = idx / N, j = idx % N;
    const float* a = A + (size_t)b * K;
    const float* w = W + (size_t)j * K;
    float s = bias ? bias[j] : 0.f;
    for (int k = 0; k < K; ++k) s += a[k] * w[k];
    C[idx] = s;
}

// ---------------------------------------------------------------------------
// Gate: g = sigmoid([cur,pooled] @ gw^T + gb); f = g*cur + (1-g)*pooled
// ---------------------------------------------------------------------------
__global__ void gate_kernel(const float* __restrict__ cur,
                            const float* __restrict__ pooled,
                            const float* __restrict__ gw,
                            const float* __restrict__ gb,
                            float* __restrict__ f)
{
    int idx = blockIdx.x * blockDim.x + threadIdx.x;
    if (idx >= BB * DMODEL) return;
    int b = idx / DMODEL, j = idx % DMODEL;
    const float* w = gw + (size_t)j * (2 * DMODEL);
    float s = gb[j];
    for (int k = 0; k < DMODEL; ++k) s += cur[b * DMODEL + k] * w[k];
    for (int k = 0; k < DMODEL; ++k) s += pooled[b * DMODEL + k] * w[DMODEL + k];
    float g = 1.0f / (1.0f + __expf(-s));
    f[idx] = g * cur[idx] + (1.0f - g) * pooled[idx];
}

// ---------------------------------------------------------------------------
extern "C" void kernel_launch(void* const* d_in, const int* in_sizes, int n_in,
                              void* d_out, int out_size, void* d_ws, size_t ws_size,
                              hipStream_t stream)
{
    const float* x       = (const float*)d_in[0];
    const float* W_proj  = (const float*)d_in[1];
    const float* b_proj  = (const float*)d_in[2];
    const float* W_in    = (const float*)d_in[3];
    const float* conv_w  = (const float*)d_in[4];
    const float* conv_b  = (const float*)d_in[5];
    const float* W_xp    = (const float*)d_in[6];
    const float* W_dt    = (const float*)d_in[7];
    const float* b_dt    = (const float*)d_in[8];
    const float* A_log   = (const float*)d_in[9];
    const float* Dp      = (const float*)d_in[10];
    const float* W_out   = (const float*)d_in[11];
    const float* mha_in_w  = (const float*)d_in[12];
    const float* mha_in_b  = (const float*)d_in[13];
    const float* mha_out_w = (const float*)d_in[14];
    const float* mha_out_b = (const float*)d_in[15];
    const float* gate_w  = (const float*)d_in[16];
    const float* gate_b  = (const float*)d_in[17];

    float* ws = (float*)d_ws;
    size_t o = 0;
    float* hbuf   = ws + o; o += (size_t)MTOT * DMODEL;       // also reused as seq
    float* uzbuf  = ws + o; o += (size_t)MTOT * 2 * DINNER;
    float* ubuf   = ws + o; o += (size_t)MTOT * DINNER;       // u, then yg in place
    float* xdbc   = ws + o; o += (size_t)MTOT * 48;
    float* dtbuf  = ws + o; o += (size_t)MTOT * DINNER;
    float* pooled = ws + o; o += BB * DMODEL;
    float* fbuf   = ws + o; o += BB * DMODEL;
    float* vbuf   = ws + o; o += BB * DMODEL;
    float* feats  = ws + o; o += 6 * BB * DMODEL;

    // 1) h = x @ W_proj^T + b_proj          [16384,1024]x[256,1024]  (aligned, bias)
    gemm_wmma_bf16<false, true, 0><<<dim3(MTOT / 128, DMODEL / 64), 256, 0, stream>>>(
        x, EMBED, W_proj, b_proj, hbuf, DMODEL, MTOT, DMODEL, EMBED);
    // 2) uz = h @ W_in^T                    [16384,256]x[1024,256]   (aligned)
    gemm_wmma_bf16<false, false, 0><<<dim3(MTOT / 128, (2 * DINNER) / 64), 256, 0, stream>>>(
        hbuf, DMODEL, W_in, nullptr, uzbuf, 2 * DINNER, MTOT, 2 * DINNER, DMODEL);
    // 3) u = silu(causal depthwise conv(u) + conv_b)
    {
        size_t total = (size_t)MTOT * DINNER;
        conv_silu_kernel<<<(unsigned)((total + 255) / 256), 256, 0, stream>>>(
            uzbuf, conv_w, conv_b, ubuf);
    }
    // 4) xdbc = u @ W_xp^T                  [16384,512]x[48,512]     (N ragged)
    gemm_wmma_bf16<true, false, 0><<<dim3(MTOT / 128, 1), 256, 0, stream>>>(
        ubuf, DINNER, W_xp, nullptr, xdbc, 48, MTOT, 48, DINNER);
    // 5) dt = softplus(xdbc[:, :16] @ W_dt^T + b_dt)   [16384,16]x[512,16] (K ragged, bias, softplus)
    gemm_wmma_bf16<true, true, 1><<<dim3(MTOT / 128, DINNER / 64), 256, 0, stream>>>(
        xdbc, 48, W_dt, b_dt, dtbuf, DINNER, MTOT, DINNER, DTRANK);
    // 6) selective scan + D skip + silu(z) gate (in place into ubuf)
    scan_kernel<<<BB * 16, 512, 0, stream>>>(dtbuf, ubuf, xdbc, uzbuf, A_log, Dp, ubuf);
    // 7) seq = yg @ W_out^T                 [16384,512]x[256,512]    (aligned)
    gemm_wmma_bf16<false, false, 0><<<dim3(MTOT / 128, DMODEL / 64), 256, 0, stream>>>(
        ubuf, DINNER, W_out, nullptr, hbuf, DMODEL, MTOT, DMODEL, DINNER);
    // 8) pooled = mean_l(seq)
    pool_kernel<<<BB, DMODEL, 0, stream>>>(hbuf, pooled);

    // 9) hierarchical head (sequential, tiny)
    static const int NC[6]  = {5, 30, 80, 200, 600, 1500};
    static const int OFF[6] = {0, 40, 280, 920, 2520, 7320};
    const float* curp = nullptr;
    for (int i = 0; i < 6; ++i) {
        const float* fp;
        if (i == 0) {
            fp = pooled;
        } else {
            gate_kernel<<<(BB * DMODEL + 255) / 256, 256, 0, stream>>>(
                curp, pooled,
                gate_w + (size_t)(i - 1) * DMODEL * 2 * DMODEL,
                gate_b + (size_t)(i - 1) * DMODEL, fbuf);
            fp = fbuf;
        }
        // v = f @ mha_in_w[i][512:768]^T + mha_in_b[i][512:768]
        small_linear_kernel<<<(BB * DMODEL + 255) / 256, 256, 0, stream>>>(
            fp, mha_in_w + (size_t)i * 3 * DMODEL * DMODEL + (size_t)2 * DMODEL * DMODEL,
            mha_in_b + (size_t)i * 3 * DMODEL + 2 * DMODEL, vbuf, DMODEL, DMODEL);
        // cur = v @ mha_out_w[i]^T + mha_out_b[i]
        float* cur = feats + (size_t)i * BB * DMODEL;
        small_linear_kernel<<<(BB * DMODEL + 255) / 256, 256, 0, stream>>>(
            vbuf, mha_out_w + (size_t)i * DMODEL * DMODEL,
            mha_out_b + (size_t)i * DMODEL, cur, DMODEL, DMODEL);
        // logits_i = cur @ Wc_i^T + bc_i
        const float* Wc = (const float*)d_in[18 + 2 * i];
        const float* bc = (const float*)d_in[19 + 2 * i];
        small_linear_kernel<<<(BB * NC[i] + 255) / 256, 256, 0, stream>>>(
            cur, Wc, bc, (float*)d_out + OFF[i], DMODEL, NC[i]);
        curp = cur;
    }
}
